// RCNNDIN_13383118094360
// MI455X (gfx1250) — matrix operand order repaired and checked by reading, not compiled, not run on hardware
//
#include <hip/hip_runtime.h>
#include <hip/hip_bf16.h>

typedef __attribute__((ext_vector_type(16))) _Float16 v16h;
typedef __attribute__((ext_vector_type(8)))  _Float16 v8h;
typedef __attribute__((ext_vector_type(8)))  float    v8f;

#define KPAD 448          // logical K: 320 (x, padded) + 128 (h)
#define XPAD 320
#define NG   512          // 4*H gates
#define HDIM 128
#define TLEN 16
#define SEQW 16           // sequences per workgroup
#define NSEQ 6528         // 128 queries + 6400 hist titles
#define WROW 320          // padded word-emb row (f16)

// ---------------- WMMA helpers ----------------

__device__ __forceinline__ v8f wmma_f16(v16h a, v16h b, v8f c) {
  return __builtin_amdgcn_wmma_f32_16x16x32_f16(
      /*neg_a=*/false, a, /*neg_b=*/false, b,
      /*c_mod=*/(short)0, c, /*reuse_a=*/false, /*reuse_b=*/false);
}

// A tile fragment (16x32 f16) from LDS, row stride `stride`.
// ISA 7.12.2 16-bit A: lane L: m=L&15, hi=L>>4;
// halves 0-7 = K[k0+8*hi ..], halves 8-15 = K[k0+16+8*hi ..].
__device__ __forceinline__ v16h load_a_frag(const _Float16* base, int stride, int k0, int lane) {
  int m  = lane & 15;
  int hi = lane >> 4;
  const _Float16* p = base + m * stride + k0 + 8 * hi;
  v8h lo = *(const v8h*)p;
  v8h hh = *(const v8h*)(p + 16);
  v16h a;
#pragma unroll
  for (int i = 0; i < 8; ++i) { a[i] = lo[i]; a[i + 8] = hh[i]; }
  return a;
}

// B fragment (32x16) from transposed weights BT[N][KPAD] f16 in global.
// Lane L: n=n0+(L&15), hi=L>>4; halves 0-7 = K[k0+8*hi ..], halves 8-15 = K[k0+16+8*hi ..].
__device__ __forceinline__ v16h load_b_frag(const _Float16* __restrict__ BT,
                                            int n0, int k0, int lane) {
  int n  = n0 + (lane & 15);
  int hi = lane >> 4;
  const _Float16* p = BT + (long)n * KPAD + k0 + 8 * hi;
  v8h lo = *(const v8h*)p;
  v8h hh = *(const v8h*)(p + 16);
  v16h b;
#pragma unroll
  for (int i = 0; i < 8; ++i) { b[i] = lo[i]; b[i + 8] = hh[i]; }
  return b;
}

// ---- CDNA5 hardware tanh (TRANS op); v_nop honors the 1-op TRANS hazard rule ----
#if defined(__has_builtin) && __has_builtin(__builtin_amdgcn_tanhf)
__device__ __forceinline__ float fast_tanh(float x) { return __builtin_amdgcn_tanhf(x); }
#else
__device__ __forceinline__ float fast_tanh(float x) {
  float r;
  asm volatile("v_tanh_f32 %0, %1\n\tv_nop" : "=v"(r) : "v"(x));
  return r;
}
#endif
__device__ __forceinline__ float sigf(float x) { return 0.5f * fast_tanh(0.5f * x) + 0.5f; }

// memory -> LDS async copy, 16B per lane (ASYNCcnt path)
__device__ __forceinline__ void async_copy16(_Float16* lds_dst, const _Float16* gsrc) {
  unsigned lds = (unsigned)(uintptr_t)lds_dst;       // flat LDS addr truncates to DS offset
  unsigned long long ga = (unsigned long long)(uintptr_t)gsrc;
  asm volatile("global_load_async_to_lds_b128 %0, %1, off"
               :: "v"(lds), "v"(ga) : "memory");
}
__device__ __forceinline__ void wait_async0() {
  asm volatile("s_wait_asynccnt 0x0" ::: "memory");
}

// ---------------- Kernel 1a: weight prep (transpose + f16 + combine) ----------------
// BT[dir][n][k]: k<300 -> Wx[k][n]; 300<=k<320 -> 0; k>=320 -> Wh[k-320][n]
__global__ void prep_weights(const float* __restrict__ Wx_f, const float* __restrict__ Wh_f,
                             const float* __restrict__ Wx_b, const float* __restrict__ Wh_b,
                             _Float16* __restrict__ BT_f, _Float16* __restrict__ BT_b) {
  int idx = blockIdx.x * 256 + threadIdx.x;
  const int per = NG * KPAD;
  if (idx >= 2 * per) return;
  int dir = idx / per;
  int rem = idx % per;
  int n = rem / KPAD, k = rem % KPAD;
  const float* Wx = dir ? Wx_b : Wx_f;
  const float* Wh = dir ? Wh_b : Wh_f;
  float v = 0.0f;
  if (k < 300)        v = Wx[k * NG + n];
  else if (k >= XPAD) v = Wh[(k - XPAD) * NG + n];
  (dir ? BT_b : BT_f)[(long)n * KPAD + k] = (_Float16)v;
}

// ---------------- Kernel 1b: word embedding -> padded f16 rows ----------------
__global__ void prep_wemb(const float* __restrict__ word_emb, _Float16* __restrict__ wembh) {
  long idx = (long)blockIdx.x * 256 + threadIdx.x;
  if (idx >= 50000L * WROW) return;
  int r = (int)(idx / WROW), c = (int)(idx % WROW);
  float v = (c < 300) ? word_emb[(long)r * 300 + c] : 0.0f;
  wembh[idx] = (_Float16)v;
}

// ------ Kernel 2: BiLSTM encoder; WMMA + overlapped async gather + reg-resident cell ---
__global__ __launch_bounds__(256) void lstm_encode(
    const int* __restrict__ query_titles, const int* __restrict__ hist_titles,
    const _Float16* __restrict__ wembh,
    const _Float16* __restrict__ BT_f, const _Float16* __restrict__ BT_b,
    const float* __restrict__ b_f, const float* __restrict__ b_b,
    float* __restrict__ enc) {
  __shared__ _Float16 X_f[2][SEQW * WROW];   // ping-pong x-region (async-filled)
  __shared__ _Float16 X_b[2][SEQW * WROW];
  __shared__ _Float16 H_f[SEQW * HDIM];      // recurrent h (f16)
  __shared__ _Float16 H_b[SEQW * HDIM];

  const int tid  = threadIdx.x;
  const int lane = tid & 31;
  const int wave = tid >> 5;
  const int nc   = lane & 15;
  const int hi   = lane >> 4;

  // zero h0
  for (int i = tid; i < SEQW * HDIM; i += 256) {
    H_f[i] = (_Float16)0.0f;
    H_b[i] = (_Float16)0.0f;
  }
  __syncthreads();

  // async gather of token embeddings for step t into X buffer `buf`:
  // 32 rows * 40 chunks of 16B = 1280 => exactly 5 per thread (no divergence)
  auto gather = [&](int t, int buf) {
    for (int ch = tid; ch < 1280; ch += 256) {
      int r = ch / 40, c = ch % 40;
      int dir = r >> 4, s = r & 15;
      int gs = blockIdx.x * SEQW + s;
      const int* ids = (gs < 128) ? (query_titles + gs * TLEN)
                                  : (hist_titles + (long)(gs - 128) * TLEN);
      int tok = dir ? (TLEN - 1 - t) : t;
      const _Float16* src = wembh + (long)ids[tok] * WROW + c * 8;
      _Float16* dst = (dir ? X_b[buf] : X_f[buf]) + s * WROW + c * 8;
      async_copy16(dst, src);
    }
  };

  gather(0, 0);
  wait_async0();
  __syncthreads();

  // per-wave register state: this wave owns cols [wave*16, wave*16+16) of every gate
  v8f c_f, c_b, hmx_f, hmx_b;
#pragma unroll
  for (int i = 0; i < 8; ++i) {
    c_f[i] = 0.0f; c_b[i] = 0.0f;
    hmx_f[i] = -3.0e38f; hmx_b[i] = -3.0e38f;
  }

  for (int t = 0; t < TLEN; ++t) {
    const int buf = t & 1;
    // prefetch next step's embeddings while this step computes
    if (t < TLEN - 1) gather(t + 1, buf ^ 1);

    // ---- gates GEMM: tiles n0 = g*128 + wave*16 for g = i,f,g,o ----
    v8f accf[4], accb[4];
#pragma unroll
    for (int g = 0; g < 4; ++g) {
      float bvf = b_f[g * HDIM + wave * 16 + nc];
      float bvb = b_b[g * HDIM + wave * 16 + nc];
#pragma unroll
      for (int i = 0; i < 8; ++i) { accf[g][i] = bvf; accb[g][i] = bvb; }
    }
    // x contribution (K = 0..319)
    for (int k0 = 0; k0 < XPAD; k0 += 32) {
      v16h af = load_a_frag(X_f[buf], WROW, k0, lane);
      v16h ab = load_a_frag(X_b[buf], WROW, k0, lane);
#pragma unroll
      for (int g = 0; g < 4; ++g) {
        int n0 = g * HDIM + wave * 16;
        v16h bf = load_b_frag(BT_f, n0, k0, lane);
        accf[g] = wmma_f16(af, bf, accf[g]);
        v16h bb = load_b_frag(BT_b, n0, k0, lane);
        accb[g] = wmma_f16(ab, bb, accb[g]);
      }
    }
    // h contribution (K = 320..447)
#pragma unroll
    for (int k0 = 0; k0 < HDIM; k0 += 32) {
      v16h af = load_a_frag(H_f, HDIM, k0, lane);
      v16h ab = load_a_frag(H_b, HDIM, k0, lane);
#pragma unroll
      for (int g = 0; g < 4; ++g) {
        int n0 = g * HDIM + wave * 16;
        v16h bf = load_b_frag(BT_f, n0, XPAD + k0, lane);
        accf[g] = wmma_f16(af, bf, accf[g]);
        v16h bb = load_b_frag(BT_b, n0, XPAD + k0, lane);
        accb[g] = wmma_f16(ab, bb, accb[g]);
      }
    }

    // ---- LSTM cell entirely in registers (elementwise on C/D layout) ----
    v8f hf, hb;
#pragma unroll
    for (int i = 0; i < 8; ++i) {
      float ccf = sigf(accf[1][i]) * c_f[i] + sigf(accf[0][i]) * fast_tanh(accf[2][i]);
      c_f[i] = ccf;
      hf[i] = sigf(accf[3][i]) * fast_tanh(ccf);
      hmx_f[i] = fmaxf(hmx_f[i], hf[i]);
      float ccb = sigf(accb[1][i]) * c_b[i] + sigf(accb[0][i]) * fast_tanh(accb[2][i]);
      c_b[i] = ccb;
      hb[i] = sigf(accb[3][i]) * fast_tanh(ccb);
      hmx_b[i] = fmaxf(hmx_b[i], hb[i]);
    }

    __syncthreads();   // all waves done reading H (and X[buf]) this step

    // write h back (row m = 8*hi + i, col wave*16 + nc)
#pragma unroll
    for (int i = 0; i < 8; ++i) {
      int m = 8 * hi + i;
      H_f[m * HDIM + wave * 16 + nc] = (_Float16)hf[i];
      H_b[m * HDIM + wave * 16 + nc] = (_Float16)hb[i];
    }
    wait_async0();     // next X buffer fully landed (own wave's asyncs)
    __syncthreads();   // h stores + async fills visible to all waves
  }

  // ---- write time-pooled output straight from registers ----
  const long gsb = (long)blockIdx.x * SEQW;
#pragma unroll
  for (int i = 0; i < 8; ++i) {
    int m = 8 * hi + i;
    enc[(gsb + m) * 256 + wave * 16 + nc]        = hmx_f[i];
    enc[(gsb + m) * 256 + HDIM + wave * 16 + nc] = hmx_b[i];
  }
}

// ---------------- Kernel 3: ans MLP (6400x128) + user MLP (128x128) ----------------
__global__ void feat_mlp(const int* __restrict__ hist_ans_sparse, const float* __restrict__ hist_ans_dense,
                         const float* __restrict__ ans_tables, const float* __restrict__ ans_W,
                         const float* __restrict__ ans_b,
                         const int* __restrict__ user_sparse, const float* __restrict__ user_dense,
                         const float* __restrict__ user_tables, const float* __restrict__ user_W,
                         const float* __restrict__ user_b,
                         float* __restrict__ h_ans, float* __restrict__ user_out) {
  long idx = (long)blockIdx.x * 256 + threadIdx.x;
  const long NH = 6400L * 128;
  if (idx < NH) {
    int r = (int)(idx >> 7), n = (int)(idx & 127);
    float acc = ans_b[n];
    const int* sp = hist_ans_sparse + (long)r * 3;
    for (int f = 0; f < 3; ++f) {
      const float* e = ans_tables + ((long)f * 1000 + sp[f]) * 64;
      const float* w = ans_W + (f * 64) * 128 + n;
      for (int k = 0; k < 64; ++k) acc += e[k] * w[k * 128];
    }
    const float* dn = hist_ans_dense + (long)r * 5;
    const float* w = ans_W + 192 * 128 + n;
    for (int k = 0; k < 5; ++k) acc += dn[k] * w[k * 128];
    h_ans[idx] = fmaxf(acc, 0.0f);
  } else if (idx < NH + 128 * 128) {
    long j = idx - NH;
    int b = (int)(j >> 7), n = (int)(j & 127);
    float acc = user_b[n];
    const int* sp = user_sparse + b * 4;
    for (int f = 0; f < 4; ++f) {
      const float* e = user_tables + ((long)f * 1000 + sp[f]) * 64;
      const float* w = user_W + (f * 64) * 128 + n;
      for (int k = 0; k < 64; ++k) acc += e[k] * w[k * 128];
    }
    const float* dn = user_dense + b * 8;
    const float* w = user_W + 256 * 128 + n;
    for (int k = 0; k < 8; ++k) acc += dn[k] * w[k * 128];
    user_out[j] = fmaxf(acc, 0.0f);
  }
}

// ---------------- Kernel 4: attention + pooling + final MLP (one block per batch) ------
__global__ __launch_bounds__(256) void attn_final(
    const float* __restrict__ enc, const float* __restrict__ h_ans,
    const float* __restrict__ user_v,
    const int* __restrict__ query_sparse, const float* __restrict__ query_dense,
    const float* __restrict__ q_tables, const int* __restrict__ hist_length,
    const float* __restrict__ att_W1, const float* __restrict__ att_b1,
    const float* __restrict__ att_W2, const float* __restrict__ att_b2,
    const float* __restrict__ fin_W1, const float* __restrict__ fin_b1,
    const float* __restrict__ fin_W2, const float* __restrict__ fin_b2,
    const float* __restrict__ fin_W3, const float* __restrict__ fin_b3,
    float* __restrict__ out) {
  __shared__ float q[388];
  __shared__ float qc[80];
  __shared__ float uv[128];
  __shared__ float scores[50];
  __shared__ float wgt[50];
  __shared__ float pooled[384];
  __shared__ float z[900];
  __shared__ float z1[256];
  __shared__ float z2[128];

  const int b   = blockIdx.x;
  const int tid = threadIdx.x;

  if (tid < 256) q[tid] = enc[(long)b * 256 + tid];
  if (tid < 64) {
    q[256 + tid] = q_tables[((long)0 * 1000 + query_sparse[b * 2 + 0]) * 64 + tid];
    q[320 + tid] = q_tables[((long)1 * 1000 + query_sparse[b * 2 + 1]) * 64 + tid];
  }
  if (tid < 4)   q[384 + tid] = query_dense[b * 4 + tid];
  if (tid < 128) uv[tid] = user_v[b * 128 + tid];
  if (tid < 50)  scores[tid] = att_b2[0];
  __syncthreads();

  if (tid < 80) {
    float acc = att_b1[tid];
    for (int k = 0; k < 388; ++k) acc += q[k] * att_W1[k * 80 + tid];
    qc[tid] = acc;
  }
  __syncthreads();

  const int len = hist_length[b];
  for (int p = tid; p < 50 * 80; p += 256) {
    int l = p / 80, j = p % 80;
    float m = (l < len) ? 1.0f : 0.0f;
    const float* he = enc + (long)(128 + b * 50 + l) * 256;
    const float* ha = h_ans + (long)(b * 50 + l) * 128;
    float acc = qc[j];
    const float* w = att_W1 + 388 * 80 + j;
    for (int k = 0; k < 256; ++k) acc += m * he[k] * w[k * 80];
    for (int k = 0; k < 128; ++k) acc += m * ha[k] * w[(256 + k) * 80];
    atomicAdd(&scores[l], fmaxf(acc, 0.0f) * att_W2[j]);
  }
  __syncthreads();

  if (tid == 0) {
    float mx = -1e30f;
    for (int l = 0; l < 50; ++l) {
      float s = (l < len) ? scores[l] : -1.0e9f;
      scores[l] = s;
      mx = fmaxf(mx, s);
    }
    float sum = 0.0f;
    for (int l = 0; l < 50; ++l) { float e = __expf(scores[l] - mx); wgt[l] = e; sum += e; }
    float inv = 1.0f / sum;
    for (int l = 0; l < 50; ++l) wgt[l] = (l < len) ? wgt[l] * inv : 0.0f;
  }
  __syncthreads();

  for (int d = tid; d < 384; d += 256) {
    float acc = 0.0f;
    for (int l = 0; l < len; ++l) {
      float h = (d < 256) ? enc[(long)(128 + b * 50 + l) * 256 + d]
                          : h_ans[(long)(b * 50 + l) * 128 + (d - 256)];
      acc += wgt[l] * h;
    }
    pooled[d] = acc;
  }
  __syncthreads();

  for (int i = tid; i < 900; i += 256)
    z[i] = (i < 388) ? q[i] : (i < 772 ? pooled[i - 388] : uv[i - 772]);
  __syncthreads();

  if (tid < 256) {
    float acc = fin_b1[tid];
    for (int k = 0; k < 900; ++k) acc += z[k] * fin_W1[k * 256 + tid];
    z1[tid] = fmaxf(acc, 0.0f);
  }
  __syncthreads();
  if (tid < 128) {
    float acc = fin_b2[tid];
    for (int k = 0; k < 256; ++k) acc += z1[k] * fin_W2[k * 128 + tid];
    z2[tid] = fmaxf(acc, 0.0f);
  }
  __syncthreads();
  if (tid == 0) {
    float acc = fin_b3[0];
    for (int k = 0; k < 128; ++k) acc += z2[k] * fin_W3[k];
    out[b] = 1.0f / (1.0f + __expf(-acc));
  }
}

// ---------------- host launcher ----------------
extern "C" void kernel_launch(void* const* d_in, const int* in_sizes, int n_in,
                              void* d_out, int out_size, void* d_ws, size_t ws_size,
                              hipStream_t stream) {
  const int*   query_titles    = (const int*)  d_in[0];
  const int*   query_sparse    = (const int*)  d_in[1];
  const float* query_dense     = (const float*)d_in[2];
  const int*   hist_titles     = (const int*)  d_in[3];
  const int*   hist_ans_sparse = (const int*)  d_in[4];
  const float* hist_ans_dense  = (const float*)d_in[5];
  const int*   hist_length     = (const int*)  d_in[6];
  const int*   user_sparse     = (const int*)  d_in[7];
  const float* user_dense      = (const float*)d_in[8];
  const float* word_emb        = (const float*)d_in[9];
  const float* Wx_f            = (const float*)d_in[10];
  const float* Wh_f            = (const float*)d_in[11];
  const float* b_f             = (const float*)d_in[12];
  const float* Wx_b            = (const float*)d_in[13];
  const float* Wh_b            = (const float*)d_in[14];
  const float* b_b             = (const float*)d_in[15];
  const float* q_tables        = (const float*)d_in[16];
  const float* ans_tables      = (const float*)d_in[17];
  const float* user_tables     = (const float*)d_in[18];
  const float* ans_W           = (const float*)d_in[19];
  const float* ans_b           = (const float*)d_in[20];
  const float* user_W          = (const float*)d_in[21];
  const float* user_b          = (const float*)d_in[22];
  const float* att_W1          = (const float*)d_in[23];
  const float* att_b1          = (const float*)d_in[24];
  const float* att_W2          = (const float*)d_in[25];
  const float* att_b2          = (const float*)d_in[26];
  const float* fin_W1          = (const float*)d_in[27];
  const float* fin_b1          = (const float*)d_in[28];
  const float* fin_W2          = (const float*)d_in[29];
  const float* fin_b2          = (const float*)d_in[30];
  const float* fin_W3          = (const float*)d_in[31];
  const float* fin_b3          = (const float*)d_in[32];

  // workspace layout
  char* ws = (char*)d_ws;
  size_t o = 0;
  _Float16* BT_f  = (_Float16*)(ws + o); o += (size_t)NG * KPAD * 2;     // 458752
  _Float16* BT_b  = (_Float16*)(ws + o); o += (size_t)NG * KPAD * 2;
  _Float16* wembh = (_Float16*)(ws + o); o += 50000L * WROW * 2;         // 32 MB
  float* enc      = (float*)(ws + o);    o += (size_t)NSEQ * 256 * 4;    // 6.7 MB
  float* h_ans    = (float*)(ws + o);    o += 6400L * 128 * 4;
  float* user_v   = (float*)(ws + o);    o += 128L * 128 * 4;
  (void)ws_size; (void)in_sizes; (void)n_in; (void)out_size;

  prep_weights<<<(2 * NG * KPAD + 255) / 256, 256, 0, stream>>>(
      Wx_f, Wh_f, Wx_b, Wh_b, BT_f, BT_b);

  prep_wemb<<<(int)((50000L * WROW + 255) / 256), 256, 0, stream>>>(word_emb, wembh);

  feat_mlp<<<(int)((6400L * 128 + 128 * 128 + 255) / 256), 256, 0, stream>>>(
      hist_ans_sparse, hist_ans_dense, ans_tables, ans_W, ans_b,
      user_sparse, user_dense, user_tables, user_W, user_b, h_ans, user_v);

  lstm_encode<<<NSEQ / SEQW, 256, 0, stream>>>(
      query_titles, hist_titles, wembh, BT_f, BT_b, b_f, b_b, enc);

  attn_final<<<128, 256, 0, stream>>>(
      enc, h_ans, user_v, query_sparse, query_dense, q_tables, hist_length,
      att_W1, att_b1, att_W2, att_b2, fin_W1, fin_b1, fin_W2, fin_b2,
      fin_W3, fin_b3, (float*)d_out);
}
